// SparseMultiHeadAttentionViT_84782654423155
// MI455X (gfx1250) — compile-verified, matrix operand order checked by reading, not run on hardware
//
#include <hip/hip_runtime.h>
#include <stdint.h>

// B=4, NQ=NK=1024, D=768, H=12, DK=64
#define BB 4
#define NQ 1024
#define NK 1024
#define DD 768
#define HH 12
#define DK 64

typedef __attribute__((ext_vector_type(16))) _Float16 v16h;
typedef __attribute__((ext_vector_type(8)))  _Float16 v8h;
typedef __attribute__((ext_vector_type(8)))  float    v8f;

// ---- WMMA wrapper: D = A(16x32 f16) * B(32x16 f16) + C(f32) ----
__device__ __forceinline__ v8f wmma32(v16h a, v16h b, v8f c) {
    return __builtin_amdgcn_wmma_f32_16x16x32_f16(
        false, a, false, b, (short)0, c, false, false);
}

// Fragment loader, f16 source. Per-lane K pattern for 16-bit A and B fragments:
// lane<16 : K = k0+{0..7}, k0+{16..23}; lane>=16: same with +8.
// caller passes k0 = kbase + 8*half already.
__device__ __forceinline__ v16h frag_f16(const _Float16* rowptr, int k0) {
    v8h c0 = *(const v8h*)(rowptr + k0);
    v8h c1 = *(const v8h*)(rowptr + k0 + 16);
    v16h r;
#pragma unroll
    for (int i = 0; i < 8; ++i) { r[i] = c0[i]; r[i + 8] = c1[i]; }
    return r;
}

// reduce across the 16 lanes of each wave half (rows live per-half in D layout)
__device__ __forceinline__ float redmax16(float x) {
    x = fmaxf(x, __shfl_xor(x, 8, 32));
    x = fmaxf(x, __shfl_xor(x, 4, 32));
    x = fmaxf(x, __shfl_xor(x, 2, 32));
    x = fmaxf(x, __shfl_xor(x, 1, 32));
    return x;
}
__device__ __forceinline__ float redsum16(float x) {
    x += __shfl_xor(x, 8, 32);
    x += __shfl_xor(x, 4, 32);
    x += __shfl_xor(x, 2, 32);
    x += __shfl_xor(x, 1, 32);
    return x;
}

// ---------------------------------------------------------------------------
// Kernel 0: bulk fp32 -> f16 conversion (one pass; GEMMs then stream f16).
// n must be a multiple of 8 (true for all tensors here).
// ---------------------------------------------------------------------------
__global__ __launch_bounds__(256) void cvt_f16_kernel(
    const float* __restrict__ in, _Float16* __restrict__ out, int n)
{
    const int stride = gridDim.x * 256 * 8;
    for (int i = (blockIdx.x * 256 + threadIdx.x) * 8; i + 8 <= n; i += stride) {
        v8f x = *(const v8f*)(in + i);
        v8h y;
#pragma unroll
        for (int j = 0; j < 8; ++j) y[j] = (_Float16)x[j];
        *(v8h*)(out + i) = y;
    }
}

// ---------------------------------------------------------------------------
// Kernel 1: q/k/v projections.  P = X @ W.T + b, scaled by zeta[dk].
// All operands pre-converted f16.  One wave = 16 rows x 64 cols (one head).
// q16,k16: f16 [B,H,N,DK].  vT16: f16 [B,H,DK,NK] (transposed for PV B-frags).
// ---------------------------------------------------------------------------
__global__ __launch_bounds__(256) void qkv_proj_kernel(
    const _Float16* __restrict__ Xq, const _Float16* __restrict__ Xk,
    const _Float16* __restrict__ Xv,
    const _Float16* __restrict__ Wq, const float* __restrict__ bq,
    const _Float16* __restrict__ Wk, const float* __restrict__ bk,
    const _Float16* __restrict__ Wv, const float* __restrict__ bv,
    const float* __restrict__ zeta,
    _Float16* __restrict__ q16, _Float16* __restrict__ k16, _Float16* __restrict__ vT16)
{
    const int lane = threadIdx.x & 31;
    const int widx = threadIdx.x >> 5;
    const int job  = blockIdx.x * 8 + widx;          // 0 .. 9215
    const int mat  = job / (256 * HH);
    const int rem  = job % (256 * HH);
    const int mt   = rem / HH;                       // 16-row tile
    const int h    = rem % HH;                       // head == 64-col group
    const int hf   = lane >> 4;
    const int lcol = lane & 15;

    const _Float16* X    = (mat == 0) ? Xq : (mat == 1) ? Xk : Xv;
    const _Float16* W    = (mat == 0) ? Wq : (mat == 1) ? Wk : Wv;
    const float*    bias = (mat == 0) ? bq : (mat == 1) ? bk : bv;

    const _Float16* arow = X + (size_t)(mt * 16 + lcol) * DD;   // A row = lane%16

    v8f vzero = {};
    v8f acc[4];
#pragma unroll
    for (int j = 0; j < 4; ++j) acc[j] = vzero;

    for (int kb = 0; kb < DD; kb += 32) {
        v16h a = frag_f16(arow, kb + 8 * hf);
#pragma unroll
        for (int j = 0; j < 4; ++j) {
            const _Float16* wrow = W + (size_t)(h * 64 + j * 16 + lcol) * DD;
            v16h bf = frag_f16(wrow, kb + 8 * hf);
            acc[j] = wmma32(a, bf, acc[j]);
        }
    }

    const int m0 = mt * 16;
#pragma unroll
    for (int j = 0; j < 4; ++j) {
        const int dk = j * 16 + lcol;
        const int n  = h * 64 + dk;
        const float z  = zeta[dk];
        const float bb = bias[n];
        if (mat == 2) {
            // vT: 8 consecutive seq values per lane -> one 16B store
            const int mA   = m0 + 8 * hf;
            const int bIdx = mA >> 10;
            const int seq0 = mA & 1023;
            v8h pack;
#pragma unroll
            for (int r = 0; r < 8; ++r)
                pack[r] = (_Float16)((acc[j][r] + bb) * z);
            *(v8h*)(vT16 + (((size_t)bIdx * HH + h) * DK + dk) * (size_t)NK + seq0) = pack;
        } else {
            _Float16* dst = (mat == 0) ? q16 : k16;
#pragma unroll
            for (int r = 0; r < 8; ++r) {
                const int m    = m0 + r + 8 * hf;
                const int bIdx = m >> 10;
                const int seq  = m & 1023;
                dst[(((size_t)bIdx * HH + h) * NQ + seq) * DK + dk] =
                    (_Float16)((acc[j][r] + bb) * z);
            }
        }
    }
}

// ---------------------------------------------------------------------------
// Kernel 2: flash attention with multiplicative+log modulation and mask.
// One wave per (b,h,16-query tile); online softmax over 32-key tiles.
// attout: f16 [B, NQ, H*DK].
// ---------------------------------------------------------------------------
__global__ __launch_bounds__(256) void attn_kernel(
    const _Float16* __restrict__ q16, const _Float16* __restrict__ k16,
    const _Float16* __restrict__ vT16, const float* __restrict__ aw,
    const uint8_t* __restrict__ mask, _Float16* __restrict__ attout)
{
    __shared__ __align__(16) _Float16 Pl[8][16][32];   // per-wave prob tile

    const int lane = threadIdx.x & 31;
    const int widx = threadIdx.x >> 5;
    const int job  = blockIdx.x * 8 + widx;            // 0 .. 3071
    const int qt   = job & 63;
    const int bh   = job >> 6;                          // b*H + h
    const int b    = bh / HH;
    const int h    = bh % HH;
    const int hf   = lane >> 4;
    const int lcol = lane & 15;
    const int qbase = qt * 16;

    // A fragments of q (fixed for whole key loop): rows = lane%16
    const _Float16* qrow = q16 + ((size_t)bh * NQ + qbase + lcol) * DK;
    const v16h aq0 = frag_f16(qrow, 0  + 8 * hf);
    const v16h aq1 = frag_f16(qrow, 32 + 8 * hf);

    const float* awbase = aw + ((size_t)bh * NQ + qbase) * (size_t)NK;
    const uint8_t* mrow = mask + (size_t)b * NK;

    float mrun[8], lrun[8];
    v8f vzero = {};
    v8f oacc[4];
#pragma unroll
    for (int r = 0; r < 8; ++r) { mrun[r] = -1e30f; lrun[r] = 0.0f; }
#pragma unroll
    for (int d = 0; d < 4; ++d) oacc[d] = vzero;

    const float scale = 0.125f;                         // 1/sqrt(DK)

    for (int kb = 0; kb < NK; kb += 32) {
        // prefetch the attention-weights stream two key-tiles ahead
        // (emits global_prefetch_b8; 16 lanes cover the 16 q rows)
        if (kb + 64 < NK)
            __builtin_prefetch(awbase + (size_t)lcol * NK + kb + 64, 0, 3);

        // ---- scores: two 16-key subtiles, K = dk (2 steps of 32) ----
        v8f s0 = vzero, s1 = vzero;
        {
            const _Float16* k0row = k16 + ((size_t)bh * NK + kb + lcol) * DK;
            const _Float16* k1row = k0row + 16 * DK;
            s0 = wmma32(aq0, frag_f16(k0row, 0  + 8 * hf), s0);
            s0 = wmma32(aq1, frag_f16(k0row, 32 + 8 * hf), s0);
            s1 = wmma32(aq0, frag_f16(k1row, 0  + 8 * hf), s1);
            s1 = wmma32(aq1, frag_f16(k1row, 32 + 8 * hf), s1);
        }
        const bool mk0 = mrow[kb + lcol]      != 0;     // true => masked (-1e9)
        const bool mk1 = mrow[kb + 16 + lcol] != 0;

#pragma unroll
        for (int r = 0; r < 8; ++r) {
            const float* awrow = awbase + (size_t)(r + 8 * hf) * NK + kb;
            const float w0 = awrow[lcol];
            const float w1 = awrow[16 + lcol];
            float x0 = s0[r] * scale * w0 + __logf(fmaxf(w0, 1e-6f));
            float x1 = s1[r] * scale * w1 + __logf(fmaxf(w1, 1e-6f));
            x0 = mk0 ? -1e9f : x0;
            x1 = mk1 ? -1e9f : x1;

            const float mx   = redmax16(fmaxf(x0, x1));
            const float mnew = fmaxf(mrun[r], mx);
            const float rs   = __expf(mrun[r] - mnew);
            const float e0   = __expf(x0 - mnew);
            const float e1   = __expf(x1 - mnew);
            lrun[r] = lrun[r] * rs + redsum16(e0 + e1);
            mrun[r] = mnew;
#pragma unroll
            for (int d = 0; d < 4; ++d) oacc[d][r] *= rs;

            // D-layout -> LDS (transpose staging for A fragment)
            Pl[widx][r + 8 * hf][lcol]      = (_Float16)e0;
            Pl[widx][r + 8 * hf][16 + lcol] = (_Float16)e1;
        }

        asm volatile("s_wait_dscnt 0" ::: "memory");

        // ---- PV: A = prob tile (16x32), B = vT rows (contiguous keys) ----
        const v16h aP = frag_f16(&Pl[widx][lcol][0], 8 * hf);
#pragma unroll
        for (int d = 0; d < 4; ++d) {
            const _Float16* vrw = vT16 + ((size_t)bh * DK + d * 16 + lcol) * (size_t)NK + kb;
            oacc[d] = wmma32(aP, frag_f16(vrw, 8 * hf), oacc[d]);
        }
    }

    // ---- finalize & store [B, NQ, H*DK] as f16 ----
    _Float16* orow = attout + ((size_t)b * NQ + qbase) * DD + h * DK;
#pragma unroll
    for (int r = 0; r < 8; ++r) {
        const float inv = 1.0f / lrun[r];
        const int m = r + 8 * hf;
#pragma unroll
        for (int d = 0; d < 4; ++d)
            orow[(size_t)m * DD + d * 16 + lcol] = (_Float16)(oacc[d][r] * inv);
    }
}

// ---------------------------------------------------------------------------
// Kernel 3: out = LN(queries + attout @ Wo.T + bo) * gamma + beta
// One block = 16 rows; 8 waves cover all 768 cols (6 subtiles each).
// ---------------------------------------------------------------------------
__global__ __launch_bounds__(256) void outproj_ln_kernel(
    const _Float16* __restrict__ attout, const _Float16* __restrict__ Wo16,
    const float* __restrict__ bo, const float* __restrict__ queries,
    const float* __restrict__ gamma, const float* __restrict__ beta,
    float* __restrict__ out)
{
    __shared__ float rowbuf[16][DD];

    const int lane = threadIdx.x & 31;
    const int widx = threadIdx.x >> 5;
    const int hf   = lane >> 4;
    const int lcol = lane & 15;
    const int m0   = blockIdx.x * 16;

    const _Float16* arow = attout + (size_t)(m0 + lcol) * DD;

    v8f vzero = {};
    v8f acc[6];
#pragma unroll
    for (int j = 0; j < 6; ++j) acc[j] = vzero;

    for (int kb = 0; kb < DD; kb += 32) {
        v16h a = frag_f16(arow, kb + 8 * hf);
#pragma unroll
        for (int j = 0; j < 6; ++j) {
            const _Float16* wrow = Wo16 + (size_t)(widx * 96 + j * 16 + lcol) * DD;
            acc[j] = wmma32(a, frag_f16(wrow, kb + 8 * hf), acc[j]);
        }
    }

    // epilogue: bias + residual -> LDS
#pragma unroll
    for (int j = 0; j < 6; ++j) {
        const int n = widx * 96 + j * 16 + lcol;
        const float bb = bo[n];
#pragma unroll
        for (int r = 0; r < 8; ++r) {
            const int m = m0 + r + 8 * hf;
            rowbuf[r + 8 * hf][n] = acc[j][r] + bb + queries[(size_t)m * DD + n];
        }
    }
    __syncthreads();

    // LayerNorm: 16 threads per row
    const int row = threadIdx.x >> 4;
    const int j16 = threadIdx.x & 15;
    float s = 0.0f, sq = 0.0f;
    for (int c = j16; c < DD; c += 16) {
        const float x = rowbuf[row][c];
        s += x; sq += x * x;
    }
#pragma unroll
    for (int m = 8; m >= 1; m >>= 1) {
        s  += __shfl_xor(s,  m, 32);
        sq += __shfl_xor(sq, m, 32);
    }
    const float mean = s * (1.0f / DD);
    const float var  = sq * (1.0f / DD) - mean * mean;
    const float rstd = rsqrtf(var + 1e-5f);
    const size_t gm  = (size_t)(m0 + row);
    for (int c = j16; c < DD; c += 16) {
        const float x = rowbuf[row][c];
        out[gm * DD + c] = (x - mean) * rstd * gamma[c] + beta[c];
    }
}

// ---------------------------------------------------------------------------
extern "C" void kernel_launch(void* const* d_in, const int* in_sizes, int n_in,
                              void* d_out, int out_size, void* d_ws, size_t ws_size,
                              hipStream_t stream)
{
    const float*   queries = (const float*)d_in[0];
    const float*   keys    = (const float*)d_in[1];
    const float*   values  = (const float*)d_in[2];
    const float*   aw      = (const float*)d_in[3];
    const uint8_t* amask   = (const uint8_t*)d_in[4];
    const float*   Wq      = (const float*)d_in[5];
    const float*   bq      = (const float*)d_in[6];
    const float*   Wk      = (const float*)d_in[7];
    const float*   bk      = (const float*)d_in[8];
    const float*   Wv      = (const float*)d_in[9];
    const float*   bv      = (const float*)d_in[10];
    const float*   Wo      = (const float*)d_in[11];
    const float*   bo      = (const float*)d_in[12];
    const float*   zeta    = (const float*)d_in[13];
    const float*   gamma   = (const float*)d_in[14];
    const float*   beta    = (const float*)d_in[15];
    float* outp = (float*)d_out;

    const size_t NELEM = (size_t)BB * HH * NQ * DK;    // 3,145,728 (= 4096*768)
    const size_t WELEM = (size_t)DD * DD;              //   589,824
    _Float16* q16    = (_Float16*)d_ws;
    _Float16* k16    = q16   + NELEM;
    _Float16* vT16   = k16   + NELEM;
    _Float16* attout = vT16  + NELEM;                  // [B*NQ, 768] f16
    _Float16* xq16   = attout + NELEM;                 // f16 copies of activations
    _Float16* xk16   = xq16  + NELEM;
    _Float16* xv16   = xk16  + NELEM;
    _Float16* wq16   = xv16  + NELEM;                  // f16 copies of weights
    _Float16* wk16   = wq16  + WELEM;
    _Float16* wv16   = wk16  + WELEM;
    _Float16* wo16   = wv16  + WELEM;

    // ---- one-shot fp32 -> f16 conversions (memory-bound, ~45 MB total) ----
    const int nAct = (int)NELEM;    // 4096*768, multiple of 2048
    const int nW   = (int)WELEM;    // 768*768,  multiple of 2048
    cvt_f16_kernel<<<nAct / 2048, 256, 0, stream>>>(queries, xq16, nAct);
    cvt_f16_kernel<<<nAct / 2048, 256, 0, stream>>>(keys,    xk16, nAct);
    cvt_f16_kernel<<<nAct / 2048, 256, 0, stream>>>(values,  xv16, nAct);
    cvt_f16_kernel<<<nW   / 2048, 256, 0, stream>>>(Wq,      wq16, nW);
    cvt_f16_kernel<<<nW   / 2048, 256, 0, stream>>>(Wk,      wk16, nW);
    cvt_f16_kernel<<<nW   / 2048, 256, 0, stream>>>(Wv,      wv16, nW);
    cvt_f16_kernel<<<nW   / 2048, 256, 0, stream>>>(Wo,      wo16, nW);

    // 3 mats * 256 m-tiles * 12 heads = 9216 wave-jobs / 8 waves per block
    qkv_proj_kernel<<<1152, 256, 0, stream>>>(
        xq16, xk16, xv16, wq16, bq, wk16, bk, wv16, bv, zeta, q16, k16, vT16);

    // 4*12*64 = 3072 wave-jobs / 8
    attn_kernel<<<384, 256, 0, stream>>>(q16, k16, vT16, aw, amask, attout);

    // 4096 rows / 16
    outproj_ln_kernel<<<256, 256, 0, stream>>>(
        attout, wo16, bo, queries, gamma, beta, outp);
}
